// AcceleratedInnerShiftTriple_74311524156021
// MI455X (gfx1250) — compile-verified, matrix operand order checked by reading, not run on hardware
//
#include <hip/hip_runtime.h>
#include <hip/hip_bf16.h>

typedef __attribute__((ext_vector_type(16))) __bf16         v16bf;
typedef __attribute__((ext_vector_type(8)))  float          v8f;
typedef __attribute__((ext_vector_type(8)))  unsigned short v8u16;
typedef __attribute__((ext_vector_type(16))) unsigned short v16u16;

constexpr int kB  = 4;
constexpr int kC  = 512;
constexpr int kC2 = 256;
constexpr int kN  = 64 * 64;          // 4096 pixels
constexpr int kQT = kN / 16;          // 256 query/candidate tiles
constexpr int kChunks = 8;            // candidate-dimension split for occupancy
constexpr int kTilesPerChunk = kQT / kChunks;  // 32 candidate tiles per chunk

__device__ __forceinline__ unsigned short f32_to_bf16_rne(float f) {
    unsigned int u = __float_as_uint(f);
    unsigned int lsb = (u >> 16) & 1u;
    u += 0x7FFFu + lsb;
    return (unsigned short)(u >> 16);
}

// Monotone packing of (score, index): larger score wins; equal score -> lower
// index wins (matches argmax first-occurrence semantics).
__device__ __forceinline__ unsigned long long pack_score_idx(float v, int idx) {
    unsigned int u = __float_as_uint(v);
    unsigned int key = (u & 0x80000000u) ? ~u : (u | 0x80000000u);
    return ((unsigned long long)key << 32) | (unsigned int)(~(unsigned int)idx);
}

// ---------------------------------------------------------------------------
// Kernel 1: out[:, 0:512, :] = x   (coalesced passthrough of former+latter)
// ---------------------------------------------------------------------------
__global__ void copy_passthrough(const float* __restrict__ x, float* __restrict__ out) {
    size_t i = (size_t)blockIdx.x * blockDim.x + threadIdx.x;
    size_t total = (size_t)kB * kC * kN;
    if (i >= total) return;
    size_t b = i / ((size_t)kC * kN);
    size_t rem = i - b * (size_t)kC * kN;
    out[b * (size_t)3 * kC2 * kN + rem] = x[i];
}

// ---------------------------------------------------------------------------
// Kernel 2: latN[(b*N+n)*C2 + c] = bf16( latter / (||latter||+eps) )
//           (pixel-major layout so WMMA fragment lanes load contiguous b128s)
//           Also zero-init the packed argmax table.
// ---------------------------------------------------------------------------
__global__ void normalize_latter(const float* __restrict__ x,
                                 unsigned short* __restrict__ latN,
                                 unsigned long long* __restrict__ best) {
    const int b = blockIdx.x / kN;
    const int n = blockIdx.x % kN;
    const int c = threadIdx.x;  // blockDim == 256 == C2

    float l = x[(((size_t)b * kC) + kC2 + c) * kN + n];

    __shared__ float red[kC2];
    red[c] = l * l;
    __syncthreads();
    for (int s = kC2 / 2; s > 0; s >>= 1) {
        if (c < s) red[c] += red[c + s];
        __syncthreads();
    }
    float inv = 1.0f / (sqrtf(red[0]) + 1e-8f);
    latN[((size_t)b * kN + n) * kC2 + c] = f32_to_bf16_rne(l * inv);
    if (c == 0) best[(size_t)b * kN + n] = 0ull;
}

// ---------------------------------------------------------------------------
// Kernel 3: WMMA bf16 Gram-matrix nearest-neighbor search.
// One wave (32 threads) per (candidate-chunk, query-tile, batch).
// A/B 16x32 bf16 fragments per documented layout:
//   lanes 0-15 hold row (M or N) = lane, K = {kb..kb+7, kb+16..kb+23}
//   lanes 16-31 hold row = lane-16, K = {kb+8..kb+15, kb+24..kb+31}
// D layout: lane -> candidate column (lane&15), acc[j] -> query row
//   j + 8*(lane>=16).
// ---------------------------------------------------------------------------
__device__ __forceinline__ v16bf load_frag(const unsigned short* __restrict__ row_k0) {
    // row_k0 already points at (row base + kbase + (laneHalf ? 8 : 0))
    v8u16 lo = *(const v8u16*)(row_k0);
    v8u16 hi = *(const v8u16*)(row_k0 + 16);
    v16u16 u;
#pragma unroll
    for (int i = 0; i < 8; ++i) { u[i] = lo[i]; u[8 + i] = hi[i]; }
    return __builtin_bit_cast(v16bf, u);
}

// Issue all 16 b128 loads for one candidate tile (no data dependence between
// them -> compiler can clause them and defer the wait).
__device__ __forceinline__ void load_btile(const unsigned short* __restrict__ row_k0,
                                           v16bf* __restrict__ bf) {
#pragma unroll
    for (int ks = 0; ks < 8; ++ks) bf[ks] = load_frag(row_k0 + ks * 32);
}

// 8 chained WMMAs over K=256, then per-lane running-argmax update.
__device__ __forceinline__ void do_tile(const v16bf* __restrict__ a,
                                        const v16bf* __restrict__ bf,
                                        int m, bool valid,
                                        float* __restrict__ bestv,
                                        int* __restrict__ bestm) {
    v8f acc = {};
#pragma unroll
    for (int ks = 0; ks < 8; ++ks) {
        acc = __builtin_amdgcn_wmma_f32_16x16x32_bf16(
            false, a[ks], false, bf[ks], (short)0, acc, false, false);
    }
    if (valid) {
#pragma unroll
        for (int j = 0; j < 8; ++j) {
            float v = acc[j];
            if (v > bestv[j]) { bestv[j] = v; bestm[j] = m; }  // strict > keeps lowest m
        }
    }
}

__global__ void __launch_bounds__(32)
nn_search(const unsigned short* __restrict__ latN,
          const float* __restrict__ mask,
          unsigned long long* __restrict__ best) {
    const int lane  = threadIdx.x;
    const int chunk = blockIdx.x;
    const int qt    = blockIdx.y;
    const int b     = blockIdx.z;
    const int n0    = qt * 16;
    const int rsel     = lane & 15;
    const int laneHalf = (lane >> 4) & 1;

    // Wave-uniform early exit: tile has no masked queries -> nothing to do.
    float mq = mask[n0 + rsel];
    if (!__any((lane < 16) && (mq >= 1.0f))) return;

    const unsigned short* base = latN + (size_t)b * kN * kC2;
    const int t0 = chunk * kTilesPerChunk;

    // Per-lane candidate-validity bitmap for the 32 tiles of this chunk
    // (loaded up-front so the main loop has zero mask-related memory waits).
    unsigned validBits = 0u;
#pragma unroll
    for (int t = 0; t < kTilesPerChunk; ++t) {
        float mc = mask[(t0 + t) * 16 + rsel];
        validBits |= (mc < 1.0f ? 1u : 0u) << t;
    }
    // Wave-OR -> per-tile "any valid candidate" mask, made scalar for uniform
    // loop control (EXEC stays all-1s around the WMMAs).
    unsigned tam = validBits;
#pragma unroll
    for (int off = 1; off < 16; off <<= 1) tam |= __shfl_xor(tam, off, 32);
    tam = __builtin_amdgcn_readfirstlane(tam);
    if (tam == 0u) return;

    // Preload full-K A fragment for the 16 queries (8 k-steps x v16bf).
    v16bf a[8];
    {
        const unsigned short* arow = base + (size_t)(n0 + rsel) * kC2 + laneHalf * 8;
#pragma unroll
        for (int ks = 0; ks < 8; ++ks) a[ks] = load_frag(arow + ks * 32);
    }

    float bestv[8];
    int   bestm[8];
#pragma unroll
    for (int j = 0; j < 8; ++j) { bestv[j] = -1e30f; bestm[j] = 0; }

    // Software-pipelined ping-pong over valid candidate tiles: issue next
    // tile's 16 loads before computing on the current buffer.
    v16bf b0[8], b1[8];
    unsigned rem = tam;
    int tc = __builtin_ctz(rem);
    rem &= rem - 1u;
    load_btile(base + (size_t)((t0 + tc) * 16 + rsel) * kC2 + laneHalf * 8, b0);

    for (;;) {
        int tn = -1;
        if (rem) {
            tn = __builtin_ctz(rem);
            rem &= rem - 1u;
            load_btile(base + (size_t)((t0 + tn) * 16 + rsel) * kC2 + laneHalf * 8, b1);
        }
        do_tile(a, b0, (t0 + tc) * 16 + rsel, (validBits >> tc) & 1u, bestv, bestm);
        if (tn < 0) break;

        tc = tn;
        tn = -1;
        if (rem) {
            tn = __builtin_ctz(rem);
            rem &= rem - 1u;
            load_btile(base + (size_t)((t0 + tn) * 16 + rsel) * kC2 + laneHalf * 8, b0);
        }
        do_tile(a, b1, (t0 + tc) * 16 + rsel, (validBits >> tc) & 1u, bestv, bestm);
        if (tn < 0) break;
        tc = tn;
    }

    // Argmax reduction across the 16 lanes of each half (candidates -> one best).
#pragma unroll
    for (int j = 0; j < 8; ++j) {
        float v = bestv[j];
        int   idx = bestm[j];
#pragma unroll
        for (int off = 1; off < 16; off <<= 1) {
            float ov = __shfl_xor(v, off, 32);
            int   oi = __shfl_xor(idx, off, 32);
            if (ov > v || (ov == v && oi < idx)) { v = ov; idx = oi; }
        }
        bestv[j] = v;
        bestm[j] = idx;
    }

    // Lanes 0 and 16 publish queries [n0..n0+7] and [n0+8..n0+15].
    if (rsel == 0) {
#pragma unroll
        for (int j = 0; j < 8; ++j) {
            if (bestv[j] > -1e29f) {
                int q = n0 + j + laneHalf * 8;
                atomicMax(&best[(size_t)b * kN + q], pack_score_idx(bestv[j], bestm[j]));
            }
        }
    }
}

// ---------------------------------------------------------------------------
// Kernel 4: shift block. out[:, 512+c, n] = flag[n] ? former[:, c, nn_idx[n]] : 0
// One block per (b, c); coalesced writes over n.
// ---------------------------------------------------------------------------
__global__ void gather_shift(const float* __restrict__ x,
                             const float* __restrict__ mask,
                             const unsigned long long* __restrict__ best,
                             float* __restrict__ out) {
    const int b = blockIdx.x / kC2;
    const int c = blockIdx.x % kC2;
    const float* former = x + (size_t)b * kC * kN + (size_t)c * kN;
    float* o = out + (size_t)b * 3 * kC2 * kN + (size_t)(2 * kC2 + c) * kN;
    for (int n = threadIdx.x; n < kN; n += blockDim.x) {
        float val = 0.0f;
        if (mask[n] >= 1.0f) {
            unsigned long long p = best[(size_t)b * kN + n];
            int idx = (int)(~(unsigned int)p);
            val = former[idx];
        }
        o[n] = val;
    }
}

extern "C" void kernel_launch(void* const* d_in, const int* in_sizes, int n_in,
                              void* d_out, int out_size, void* d_ws, size_t ws_size,
                              hipStream_t stream) {
    (void)in_sizes; (void)n_in; (void)out_size; (void)ws_size;
    const float* x    = (const float*)d_in[0];
    const float* mask = (const float*)d_in[1];
    float* out        = (float*)d_out;

    // Workspace: bf16 normalized latter (8 MB), then packed argmax table (128 KB).
    unsigned short* latN = (unsigned short*)d_ws;
    unsigned long long* best =
        (unsigned long long*)((char*)d_ws + (size_t)kB * kN * kC2 * sizeof(unsigned short));

    size_t total = (size_t)kB * kC * kN;
    copy_passthrough<<<(unsigned)((total + 255) / 256), 256, 0, stream>>>(x, out);
    normalize_latter<<<kB * kN, kC2, 0, stream>>>(x, latN, best);
    dim3 grid(kChunks, kQT, kB);
    nn_search<<<grid, 32, 0, stream>>>(latN, mask, best);
    gather_shift<<<kB * kC2, 256, 0, stream>>>(x, mask, best, out);
}
